// testModel_1717986919018
// MI455X (gfx1250) — matrix operand order repaired and checked by reading, not compiled, not run on hardware
//
#include <hip/hip_runtime.h>
#include <hip/hip_bf16.h>
#include <math.h>

typedef __attribute__((ext_vector_type(16))) __bf16 v16bf;
typedef __attribute__((ext_vector_type(8)))  __bf16 v8bf;
typedef __attribute__((ext_vector_type(8)))  float  v8f;

#define CFEAT 128

// ---------------- small helpers ----------------

__device__ inline float atomicMaxFloat(float* addr, float value) {
    // works for mixed signs when addr initialized to -inf
    if (value >= 0.0f)
        return __int_as_float(atomicMax((int*)addr, __float_as_int(value)));
    else
        return __uint_as_float(atomicMin((unsigned int*)addr, __float_as_uint(value)));
}

__device__ inline v16bf load_a_frag_f32(const float* rowp, int kb) {
    // elements 0..15 -> K = kb .. kb+15
    float4 a0 = *(const float4*)(rowp + kb);
    float4 a1 = *(const float4*)(rowp + kb + 4);
    float4 a2 = *(const float4*)(rowp + kb + 8);
    float4 a3 = *(const float4*)(rowp + kb + 12);
    v16bf f;
    f[0]  = (__bf16)a0.x; f[1]  = (__bf16)a0.y; f[2]  = (__bf16)a0.z; f[3]  = (__bf16)a0.w;
    f[4]  = (__bf16)a1.x; f[5]  = (__bf16)a1.y; f[6]  = (__bf16)a1.z; f[7]  = (__bf16)a1.w;
    f[8]  = (__bf16)a2.x; f[9]  = (__bf16)a2.y; f[10] = (__bf16)a2.z; f[11] = (__bf16)a2.w;
    f[12] = (__bf16)a3.x; f[13] = (__bf16)a3.y; f[14] = (__bf16)a3.z; f[15] = (__bf16)a3.w;
    return f;
}

__device__ inline v16bf load_b_frag_lds(const __bf16* base) {
    const v8bf* p = (const v8bf*)base;
    v8bf lo = p[0];
    v8bf hi = p[1];
    v16bf f;
#pragma unroll
    for (int i = 0; i < 8; ++i) { f[i] = lo[i]; f[i + 8] = hi[i]; }
    return f;
}

// ---------------- init / fold ----------------

__global__ void fill_kernel(float* p, float v, int n) {
    int i = blockIdx.x * 256 + threadIdx.x;
    if (i < n) p[i] = v;
}

// u[k] = sum_j We[k][j] * we[j];  c = be . we + att_b
__global__ void fold_kernel(const float* __restrict__ We, const float* __restrict__ be,
                            const float* __restrict__ attw, const float* __restrict__ attb,
                            float* __restrict__ u, float* __restrict__ c) {
    const float* we = attw + 2 * CFEAT;
    int k = threadIdx.x; // 128 threads
    float s = 0.0f;
    for (int j = 0; j < CFEAT; ++j) s += We[(size_t)k * CFEAT + j] * we[j];
    u[k] = s;
    if (k == 0) {
        float cc = 0.0f;
        for (int j = 0; j < CFEAT; ++j) cc += be[j] * we[j];
        c[0] = cc + attb[0];
    }
}

// ---------------- WMMA GEMM: out[M,128] = in[M,128] @ W[128,128] + b ----------------

__global__ void gemm128_kernel(const float* __restrict__ A, const float* __restrict__ W,
                               const float* __restrict__ bias, float* __restrict__ out, int M) {
    __shared__ __bf16 Wt[CFEAT][CFEAT + 8]; // transposed [N][K], padded vs bank conflicts

    // stage weights -> LDS as bf16 transposed
    for (int idx = threadIdx.x; idx < CFEAT * CFEAT; idx += 256) {
        int k = idx >> 7;
        int n = idx & 127;
        Wt[n][k] = (__bf16)W[(size_t)k * CFEAT + n];
    }
    __syncthreads();

    const int lane = threadIdx.x & 31;
    const int wave = threadIdx.x >> 5;
    const int mbase = blockIdx.x * 128 + wave * 16;

    int mrow = mbase + (lane & 15);
    int arow = mrow < M ? mrow : (M - 1);
    const float* rowp = A + (size_t)arow * CFEAT;

    float bvals[8];
#pragma unroll
    for (int nt = 0; nt < 8; ++nt) bvals[nt] = bias[nt * 16 + (lane & 15)];

    v8f acc[8];
#pragma unroll
    for (int nt = 0; nt < 8; ++nt) acc[nt] = (v8f)(0.0f);

#pragma unroll
    for (int kk = 0; kk < CFEAT; kk += 32) {
        int kb = kk + ((lane & 16) ? 16 : 0);
        v16bf afrag = load_a_frag_f32(rowp, kb);
        int kcol = kk + ((lane & 16) ? 16 : 0);
#pragma unroll
        for (int nt = 0; nt < 8; ++nt) {
            int n = nt * 16 + (lane & 15);
            v16bf bfrag = load_b_frag_lds(&Wt[n][kcol]);
            acc[nt] = __builtin_amdgcn_wmma_f32_16x16x32_bf16(
                false, afrag, false, bfrag, (short)0, acc[nt], false, false);
        }
    }

#pragma unroll
    for (int nt = 0; nt < 8; ++nt) {
        int coln = nt * 16 + (lane & 15);
#pragma unroll
        for (int r = 0; r < 8; ++r) {
            int row = mbase + r + ((lane & 16) ? 8 : 0);
            if (row < M) out[(size_t)row * CFEAT + coln] = acc[nt][r] + bvals[nt];
        }
    }
}

// ---------------- per-node dots: a_wi[n]=x[n].wi, a_wj[n]=x[n].wj ----------------

__global__ void node_dot_kernel(const float* __restrict__ x, const float* __restrict__ attw,
                                float* __restrict__ awi, float* __restrict__ awj, int N) {
    int lane = threadIdx.x & 31;
    int n = blockIdx.x * 8 + (threadIdx.x >> 5);
    if (n >= N) return;
    const float* row = x + (size_t)n * CFEAT;
    float4 xv  = *(const float4*)(row + lane * 4);
    float4 wi4 = *(const float4*)(attw + lane * 4);
    float4 wj4 = *(const float4*)(attw + CFEAT + lane * 4);
    float pi = xv.x * wi4.x + xv.y * wi4.y + xv.z * wi4.z + xv.w * wi4.w;
    float pj = xv.x * wj4.x + xv.y * wj4.y + xv.z * wj4.z + xv.w * wj4.w;
#pragma unroll
    for (int m = 16; m >= 1; m >>= 1) {
        pi += __shfl_xor(pi, m);
        pj += __shfl_xor(pj, m);
    }
    if (lane == 0) { awi[n] = pi; awj[n] = pj; }
}

// ---------------- per-edge logit + leaky relu + segment max ----------------

__global__ void edge_logit_kernel(const float* __restrict__ emb, const int* __restrict__ src,
                                  const int* __restrict__ dst, const float* __restrict__ u,
                                  const float* __restrict__ cptr,
                                  const float* __restrict__ a_wj_src,
                                  const float* __restrict__ a_wi_dst,
                                  float* __restrict__ alpha_out, float* __restrict__ mmax, int E) {
    __shared__ float su[CFEAT];
    if (threadIdx.x < CFEAT) su[threadIdx.x] = u[threadIdx.x];
    __syncthreads();
    int e = blockIdx.x * 8 + (threadIdx.x >> 5);
    if (e >= E) return;
    int lane = threadIdx.x & 31;
    const float* row = emb + (size_t)e * CFEAT;
    float4 xv = *(const float4*)(row + lane * 4);
    float4 uv = *(const float4*)(&su[lane * 4]);
    float p = xv.x * uv.x + xv.y * uv.y + xv.z * uv.z + xv.w * uv.w;
#pragma unroll
    for (int m = 16; m >= 1; m >>= 1) p += __shfl_xor(p, m);
    if (lane == 0) {
        int d = dst[e];
        float logit = p + cptr[0] + a_wi_dst[d] + a_wj_src[src[e]];
        float alpha = logit > 0.0f ? logit : 0.2f * logit; // leaky_relu(0.2)
        alpha_out[e] = alpha;
        atomicMaxFloat(&mmax[d], alpha);
    }
}

// ---------------- per-edge exp + segment sum ----------------

__global__ void exp_sum_kernel(const float* __restrict__ alpha, const int* __restrict__ dst,
                               const float* __restrict__ mmax, float* __restrict__ ssum, int E) {
    int e = blockIdx.x * 256 + threadIdx.x;
    if (e >= E) return;
    int d = dst[e];
    atomicAdd(&ssum[d], __expf(alpha[e] - mmax[d]));
}

// ---------------- z[n,:] = relu(x[n,:] * s/(s+1e-16)) in place ----------------

__global__ void scale_relu_kernel(float* __restrict__ z, const float* __restrict__ ssum, int total) {
    int i = blockIdx.x * 256 + threadIdx.x;
    if (i >= total) return;
    int n = i >> 7;
    float s = ssum[n];
    float f = s / (s + 1e-16f);
    float v = z[i] * f;
    z[i] = v > 0.0f ? v : 0.0f;
}

// ---------------- fused final MLP with gathered rows ----------------
// h = relu(concat(zg[row], zd[col]) @ W1 + b1) ; pred = h @ W2 + b2

__global__ void pred_kernel(const float* __restrict__ zg, const float* __restrict__ zd,
                            const int* __restrict__ rows, const int* __restrict__ cols,
                            const float* __restrict__ W1, const float* __restrict__ b1v,
                            const float* __restrict__ W2, const float* __restrict__ b2,
                            float* __restrict__ pred, int EL) {
    __shared__ __bf16 Wt[CFEAT][CFEAT + 8]; // one 128-K half of W1, transposed [N][K]

    const int lane = threadIdx.x & 31;
    const int wave = threadIdx.x >> 5;
    const int mbase = blockIdx.x * 128 + wave * 16;

    float b1vals[8], w2vals[8];
#pragma unroll
    for (int nt = 0; nt < 8; ++nt) {
        int nn = nt * 16 + (lane & 15);
        b1vals[nt] = b1v[nn];
        w2vals[nt] = W2[nn];
    }

    v8f acc[8];
#pragma unroll
    for (int nt = 0; nt < 8; ++nt) acc[nt] = (v8f)(0.0f);

    int m = mbase + (lane & 15);
    int em = m < EL ? m : (EL - 1);

    for (int half = 0; half < 2; ++half) {
        __syncthreads(); // previous-phase LDS reads done
        for (int idx = threadIdx.x; idx < CFEAT * CFEAT; idx += 256) {
            int k = idx >> 7;
            int n = idx & 127;
            Wt[n][k] = (__bf16)W1[(size_t)(half * CFEAT + k) * CFEAT + n];
        }
        __syncthreads();

        const float* zsrc = half ? zd : zg;
        const int*   iarr = half ? cols : rows;
        const float* rowp = zsrc + (size_t)iarr[em] * CFEAT;

#pragma unroll
        for (int kk = 0; kk < CFEAT; kk += 32) {
            int kb = kk + ((lane & 16) ? 16 : 0);
            v16bf afrag = load_a_frag_f32(rowp, kb);
            int kcol = kk + ((lane & 16) ? 16 : 0);
#pragma unroll
            for (int nt = 0; nt < 8; ++nt) {
                int n = nt * 16 + (lane & 15);
                v16bf bfrag = load_b_frag_lds(&Wt[n][kcol]);
                acc[nt] = __builtin_amdgcn_wmma_f32_16x16x32_bf16(
                    false, afrag, false, bfrag, (short)0, acc[nt], false, false);
            }
        }
    }

    // epilogue: relu(h) . W2, reduce across the 16-lane half groups
    float bb = b2[0];
#pragma unroll
    for (int r = 0; r < 8; ++r) {
        float t = 0.0f;
#pragma unroll
        for (int nt = 0; nt < 8; ++nt) {
            float h = acc[nt][r] + b1vals[nt];
            h = h > 0.0f ? h : 0.0f;
            t += h * w2vals[nt];
        }
        t += __shfl_xor(t, 1);
        t += __shfl_xor(t, 2);
        t += __shfl_xor(t, 4);
        t += __shfl_xor(t, 8);
        if ((lane & 15) == 0) {
            int row = mbase + r + ((lane & 16) ? 8 : 0);
            if (row < EL) pred[row] = t + bb;
        }
    }
}

// ---------------- host launch ----------------

extern "C" void kernel_launch(void* const* d_in, const int* in_sizes, int n_in,
                              void* d_out, int out_size, void* d_ws, size_t ws_size,
                              hipStream_t stream) {
    const int Cc = CFEAT;
    const int NG = in_sizes[0] / Cc;
    const int ND = in_sizes[1] / Cc;
    const int E  = in_sizes[4] / Cc;
    const int EL = in_sizes[6] / 2;

    const float* x_gene   = (const float*)d_in[0];
    const float* x_dis    = (const float*)d_in[1];
    const int*   ei_g2d   = (const int*)d_in[2]; // [0:E) src(gene), [E:2E) dst(disease)
    const int*   ei_d2g   = (const int*)d_in[3]; // [0:E) src(disease), [E:2E) dst(gene)
    const float* emb_g2d  = (const float*)d_in[4];
    const float* emb_d2g  = (const float*)d_in[5];
    const int*   eli      = (const int*)d_in[6]; // [0:EL) gene row, [EL:2EL) disease col
    const float* Wn_g     = (const float*)d_in[7];
    const float* bn_g     = (const float*)d_in[8];
    const float* Wn_d     = (const float*)d_in[9];
    const float* bn_d     = (const float*)d_in[10];
    const float* We_g2d   = (const float*)d_in[11];
    const float* be_g2d   = (const float*)d_in[12];
    const float* We_d2g   = (const float*)d_in[13];
    const float* be_d2g   = (const float*)d_in[14];
    const float* att_w    = (const float*)d_in[15];
    const float* att_b    = (const float*)d_in[16];
    const float* W1       = (const float*)d_in[17];
    const float* b1       = (const float*)d_in[18];
    const float* W2       = (const float*)d_in[19];
    const float* b2       = (const float*)d_in[20];

    // workspace layout
    float* ws = (float*)d_ws;
    float* a_wi_g = ws;            ws += NG;
    float* a_wj_g = ws;            ws += NG;
    float* a_wi_d = ws;            ws += ND;
    float* a_wj_d = ws;            ws += ND;
    float* m_g2d  = ws;            ws += ND; // segment max over disease dst
    float* s_g2d  = ws;            ws += ND;
    float* m_d2g  = ws;            ws += NG; // segment max over gene dst
    float* s_d2g  = ws;            ws += NG;
    float* al_g2d = ws;            ws += E;
    float* al_d2g = ws;            ws += E;
    float* u_g2d  = ws;            ws += Cc;
    float* u_d2g  = ws;            ws += Cc;
    float* c_g2d  = ws;            ws += 1;
    float* c_d2g  = ws;            ws += 1;

    float* out  = (float*)d_out;
    float* pred = out;
    float* zg   = out + EL;                 // z_gene region (holds x_g first, then scaled)
    float* zd   = zg + (size_t)NG * Cc;     // z_disease region

    const float NEG_INF = -INFINITY;

    // 1) init segment stats
    fill_kernel<<<(ND + 255) / 256, 256, 0, stream>>>(m_g2d, NEG_INF, ND);
    fill_kernel<<<(ND + 255) / 256, 256, 0, stream>>>(s_g2d, 0.0f, ND);
    fill_kernel<<<(NG + 255) / 256, 256, 0, stream>>>(m_d2g, NEG_INF, NG);
    fill_kernel<<<(NG + 255) / 256, 256, 0, stream>>>(s_d2g, 0.0f, NG);

    // 2) fold edge-weight matrices through attention vector
    fold_kernel<<<1, 128, 0, stream>>>(We_g2d, be_g2d, att_w, att_b, u_g2d, c_g2d);
    fold_kernel<<<1, 128, 0, stream>>>(We_d2g, be_d2g, att_w, att_b, u_d2g, c_d2g);

    // 3) node feature transforms (WMMA GEMM) -> straight into z output slots
    gemm128_kernel<<<(NG + 127) / 128, 256, 0, stream>>>(x_gene, Wn_g, bn_g, zg, NG);
    gemm128_kernel<<<(ND + 127) / 128, 256, 0, stream>>>(x_dis,  Wn_d, bn_d, zd, ND);

    // 4) per-node attention dots
    node_dot_kernel<<<(NG + 7) / 8, 256, 0, stream>>>(zg, att_w, a_wi_g, a_wj_g, NG);
    node_dot_kernel<<<(ND + 7) / 8, 256, 0, stream>>>(zd, att_w, a_wi_d, a_wj_d, ND);

    // 5) per-edge logits + segment max
    // g2d: x_dst = disease -> a_wi_d[dst], x_src = gene -> a_wj_g[src]
    edge_logit_kernel<<<(E + 7) / 8, 256, 0, stream>>>(emb_g2d, ei_g2d, ei_g2d + E, u_g2d, c_g2d,
                                                       a_wj_g, a_wi_d, al_g2d, m_g2d, E);
    // d2g: x_dst = gene -> a_wi_g[dst], x_src = disease -> a_wj_d[src]
    edge_logit_kernel<<<(E + 7) / 8, 256, 0, stream>>>(emb_d2g, ei_d2g, ei_d2g + E, u_d2g, c_d2g,
                                                       a_wj_d, a_wi_g, al_d2g, m_d2g, E);

    // 6) exp + segment sums
    exp_sum_kernel<<<(E + 255) / 256, 256, 0, stream>>>(al_g2d, ei_g2d + E, m_g2d, s_g2d, E);
    exp_sum_kernel<<<(E + 255) / 256, 256, 0, stream>>>(al_d2g, ei_d2g + E, m_d2g, s_d2g, E);

    // 7) z = relu(x_dst * s/(s+eps))  (softmax-sum collapse; empty segments -> 0)
    scale_relu_kernel<<<(ND * Cc + 255) / 256, 256, 0, stream>>>(zd, s_g2d, ND * Cc);
    scale_relu_kernel<<<(NG * Cc + 255) / 256, 256, 0, stream>>>(zg, s_d2g, NG * Cc);

    // 8) fused final MLP on gathered label edges (WMMA)
    pred_kernel<<<(EL + 127) / 128, 256, 0, stream>>>(zg, zd, eli, eli + EL, W1, b1, W2, b2,
                                                      pred, EL);
}
